// NewLLMFlexible_71768903516585
// MI455X (gfx1250) — matrix operand order repaired and checked by reading, not compiled, not run on hardware
//
#include <hip/hip_runtime.h>

// ---------------------------------------------------------------------------
// CDNA5 (gfx1250, wave32) implementation of the reference "NewLLMFlexible".
//
// Roofline: ~1.69 TFLOP total, but unavoidable HBM traffic is ~1.35 GB
// (weights 827 MB read once + logits 524 MB written) -> ~58 us @ 23.3 TB/s.
// WMMA bf16 compute is orders of magnitude faster than that, so the design
// goal is pure streaming efficiency:
//   - v_wmma_f32_16x16x32_bf16 as the MAC engine (fp32->bf16 on the fly)
//   - A tiles copied global->LDS with the gfx1250 async-DMA path
//     (global_load_async_to_lds_b128, ASYNCcnt) - no VGPR round trip
//   - W tiles converted fp32->bf16 and stored transposed [n][k] with packed
//     ds_store_b64 so B fragments are contiguous ds_load_b128
//   - blockIdx.x = M (fast dim) so the 32 M-blocks sharing one weight panel
//     run concurrently and weights stream from HBM exactly once via L2.
// ---------------------------------------------------------------------------

typedef __attribute__((ext_vector_type(16))) __bf16 v16bf;
typedef __attribute__((ext_vector_type(8)))  __bf16 v8bf;
typedef __attribute__((ext_vector_type(4)))  __bf16 v4bf;
typedef __attribute__((ext_vector_type(8)))  float  v8f;

#define ACT_NONE 0
#define ACT_RELU 1

// ---------------------------------------------------------------------------
// Tiled WMMA GEMM:  Cout[M,N] = act(A_bf16[M,K] @ W_f32[K,N] + bias)
//  - 128x128 tile per 256-thread block (8 wave32 waves)
//  - wave grid 4(M) x 2(N): each wave -> 2x4 = 8 16x16 f32 accumulators
//  - K stepped by 32 (one bf16 WMMA per accumulator per step)
// ---------------------------------------------------------------------------
template <int ACT, typename OutT>
__global__ __launch_bounds__(256) void gemm_wmma_kernel(
    const __bf16* __restrict__ A, int lda,
    const float*  __restrict__ W, int ldw,
    const float*  __restrict__ bias,
    OutT* __restrict__ Cout, int ldc, int K)
{
    __shared__ __bf16 Alds[128][40];   // [m][k], stride 40 keeps 16B alignment
    __shared__ __bf16 Blds[128][40];   // [n][k] (transposed on store)

    const int mblk = blockIdx.x * 128;   // M is the fast grid dim (L2 reuse)
    const int nblk = blockIdx.y * 128;
    const int tid  = threadIdx.x;
    const int lane = tid & 31;
    const int wave = tid >> 5;
    const int wmr  = (wave & 3) * 32;   // wave row base within tile
    const int wnc  = (wave >> 2) * 64;  // wave col base within tile

    v8f acc[2][4];
#pragma unroll
    for (int i = 0; i < 2; ++i)
#pragma unroll
        for (int j = 0; j < 4; ++j)
#pragma unroll
            for (int v = 0; v < 8; ++v) acc[i][j][v] = 0.0f;

    // ---- staging assignments (256 threads) ----
    // A: thread copies 16 contiguous bf16 (one row segment) via async DMA
    const int am    = tid >> 1;            // 0..127  : A row
    const int akseg = (tid & 1) * 16;      // 0 / 16  : A k segment
    // LDS byte address of this thread's A staging slot (low 32 bits of the
    // generic pointer are the LDS byte offset per the ISA aperture rules).
    const unsigned a_lds = (unsigned)(unsigned long long)&Alds[am][akseg];

    // W: thread owns a 4k x 4n micro-tile -> 4 coalesced float4 loads,
    //    4 packed ds_store_b64 into the transposed [n][k] layout.
    const int bkq = (tid >> 5) * 4;        // k base 0..28 (wave-uniform)
    const int bnq = (tid & 31) * 4;        // n base 0..124

    // ---- fragment-load lane decomposition (ISA 16-bit A/B layouts) ----
    const int arow = lane & 15;
    const int akb  = (lane >> 4) * 8;      // lanes 16-31 hold K=8..15 / 24..31
    const int bn   = lane & 15;
    const int bkb  = (lane >> 4) * 8;      // lanes 16-31 hold K=8..15 / 24..31

    for (int k0 = 0; k0 < K; k0 += 32) {
        // ---- stage A tile: async global->LDS DMA (ASYNCcnt) ----
        const __bf16* ap = A + (size_t)(mblk + am) * lda + (k0 + akseg);
        asm volatile("global_load_async_to_lds_b128 %0, %1, off"
                     :: "v"(a_lds), "v"(ap) : "memory");
        asm volatile("global_load_async_to_lds_b128 %0, %1, off offset:16"
                     :: "v"(a_lds), "v"(ap) : "memory");

        // ---- stage B tile: fp32 global -> bf16 LDS, transposed ----
        const float* wp = W + (size_t)(k0 + bkq) * ldw + (nblk + bnq);
        if (k0 + 32 < K) __builtin_prefetch(wp + (size_t)32 * ldw, 0, 1);
        float wv[4][4];
#pragma unroll
        for (int kk = 0; kk < 4; ++kk) {
            float4 w4 = *(const float4*)(wp + (size_t)kk * ldw);
            wv[kk][0] = w4.x; wv[kk][1] = w4.y; wv[kk][2] = w4.z; wv[kk][3] = w4.w;
        }
#pragma unroll
        for (int nn = 0; nn < 4; ++nn) {
            v4bf pk;
            pk[0] = (__bf16)wv[0][nn];
            pk[1] = (__bf16)wv[1][nn];
            pk[2] = (__bf16)wv[2][nn];
            pk[3] = (__bf16)wv[3][nn];
            *(v4bf*)&Blds[bnq + nn][bkq] = pk;   // 8B packed store along k
        }

        // async A copies + our ds stores must land before the barrier
        asm volatile("s_wait_asynccnt 0x0" ::: "memory");
        __syncthreads();

        // ---- fragments ----
        v16bf afrag[2];
        v16bf bfrag[4];
#pragma unroll
        for (int i = 0; i < 2; ++i) {
            const __bf16* p = &Alds[wmr + i * 16 + arow][akb];
            v8bf lo = *(const v8bf*)p;          // K = akb .. akb+7
            v8bf hi = *(const v8bf*)(p + 16);   // K = akb+16 .. akb+23
            afrag[i] = __builtin_shufflevector(lo, hi,
                0,1,2,3,4,5,6,7,8,9,10,11,12,13,14,15);
        }
#pragma unroll
        for (int j = 0; j < 4; ++j) {
            const __bf16* p = &Blds[wnc + j * 16 + bn][bkb];
            v8bf lo = *(const v8bf*)p;          // K = bkb .. bkb+7
            v8bf hi = *(const v8bf*)(p + 16);   // K = bkb+16 .. bkb+23
            bfrag[j] = __builtin_shufflevector(lo, hi,
                0,1,2,3,4,5,6,7,8,9,10,11,12,13,14,15);
        }

        // ---- 8 WMMAs per wave per K step ----
#pragma unroll
        for (int i = 0; i < 2; ++i)
#pragma unroll
            for (int j = 0; j < 4; ++j)
                acc[i][j] = __builtin_amdgcn_wmma_f32_16x16x32_bf16(
                    false, afrag[i], false, bfrag[j],
                    (short)0, acc[i][j], false, false);
        __syncthreads();
    }

    // ---- epilogue: bias (+ReLU) and store (bf16 or f32) ----
    const int rsub = (lane >> 4) * 8;   // lanes 16-31 hold rows M+8..M+15
#pragma unroll
    for (int i = 0; i < 2; ++i) {
#pragma unroll
        for (int j = 0; j < 4; ++j) {
            const int col = nblk + wnc + j * 16 + (lane & 15);
            const float bv = bias[col];
#pragma unroll
            for (int v = 0; v < 8; ++v) {
                const int row = mblk + wmr + i * 16 + rsub + v;
                float x = acc[i][j][v] + bv;
                if (ACT == ACT_RELU) x = fmaxf(x, 0.0f);
                Cout[(size_t)row * ldc + col] = (OutT)x;
            }
        }
    }
}

// ---------------------------------------------------------------------------
// Block-wide sum reduction of two values (256 threads).
// ---------------------------------------------------------------------------
__device__ __forceinline__ void block_reduce2(float& a, float& b,
                                              float* s1, float* s2)
{
    const int t = threadIdx.x;
    s1[t] = a; s2[t] = b;
    __syncthreads();
#pragma unroll
    for (int off = 128; off > 0; off >>= 1) {
        if (t < off) { s1[t] += s1[t + off]; s2[t] += s2[t + off]; }
        __syncthreads();
    }
    a = s1[0]; b = s2[0];
    __syncthreads();
}

// ---------------------------------------------------------------------------
// embed lookup + LayerNorm -> bf16 into left half of xcat[tok][0..1023];
// zero-init ctx (right half). One block per token, 256 thr x 4 elems.
// ---------------------------------------------------------------------------
__global__ __launch_bounds__(256) void embed_ln_kernel(
    const int* __restrict__ ids, const float* __restrict__ table,
    const float* __restrict__ g, const float* __restrict__ b,
    __bf16* __restrict__ xcat)
{
    __shared__ float s1[256], s2[256];
    const int tok = blockIdx.x;
    const int t   = threadIdx.x;
    const float* src = table + (size_t)ids[tok] * 1024;

    float4 v4 = *(const float4*)(src + t * 4);
    float x[4] = {v4.x, v4.y, v4.z, v4.w};
    float s = x[0] + x[1] + x[2] + x[3];
    float q = x[0]*x[0] + x[1]*x[1] + x[2]*x[2] + x[3]*x[3];
    block_reduce2(s, q, s1, s2);

    const float mean = s * (1.0f / 1024.0f);
    const float var  = q * (1.0f / 1024.0f) - mean * mean;
    const float inv  = rsqrtf(var + 1e-5f);

    __bf16* dst = xcat + (size_t)tok * 2048;
#pragma unroll
    for (int k = 0; k < 4; ++k) {
        const int c = t * 4 + k;
        dst[c]        = (__bf16)((x[k] - mean) * inv * g[c] + b[c]);
        dst[1024 + c] = (__bf16)0.0f;   // ctx = 0
    }
}

// ---------------------------------------------------------------------------
// ctx = LN(sigmoid(pf)*ctx + sigmoid(pi)*tanh(pd)) -> bf16 back into the
// right half of xcat. One block per token, 256 thr x 4 elems.
// ---------------------------------------------------------------------------
__global__ __launch_bounds__(256) void ctx_ln_kernel(
    const float* __restrict__ pd, const float* __restrict__ pf,
    const float* __restrict__ pi_,
    const float* __restrict__ cg, const float* __restrict__ cb,
    __bf16* __restrict__ xcat)
{
    __shared__ float s1[256], s2[256];
    const int tok = blockIdx.x;
    const int t   = threadIdx.x;
    const size_t o = (size_t)tok * 1024 + t * 4;

    float4 d4 = *(const float4*)(pd  + o);
    float4 f4 = *(const float4*)(pf  + o);
    float4 i4 = *(const float4*)(pi_ + o);
    __bf16* cptr = xcat + (size_t)tok * 2048 + 1024 + t * 4;

    float dv[4] = {d4.x, d4.y, d4.z, d4.w};
    float fv[4] = {f4.x, f4.y, f4.z, f4.w};
    float iv[4] = {i4.x, i4.y, i4.z, i4.w};

    float v[4];
#pragma unroll
    for (int k = 0; k < 4; ++k) {
        const float ctxo = (float)cptr[k];
        const float fg = 1.0f / (1.0f + __expf(-fv[k]));
        const float ig = 1.0f / (1.0f + __expf(-iv[k]));
        v[k] = fg * ctxo + ig * tanhf(dv[k]);
    }
    float s = v[0] + v[1] + v[2] + v[3];
    float q = v[0]*v[0] + v[1]*v[1] + v[2]*v[2] + v[3]*v[3];
    block_reduce2(s, q, s1, s2);

    const float mean = s * (1.0f / 1024.0f);
    const float var  = q * (1.0f / 1024.0f) - mean * mean;
    const float inv  = rsqrtf(var + 1e-5f);
#pragma unroll
    for (int k = 0; k < 4; ++k) {
        const int c = t * 4 + k;
        cptr[k] = (__bf16)((v[k] - mean) * inv * cg[c] + cb[c]);
    }
}

// ---------------------------------------------------------------------------
// Host-side orchestration.
// Workspace layout (96 MB for N=4096 tokens):
//   xcat : [N][2048] bf16   (x | ctx, concat is free)              16 MB
//   R1   : h1 [N][4096] bf16  ALIASED with g0/g1/g2 f32 [N][1024]  48 MB
//   h    : [N][4096] bf16                                          32 MB
// ---------------------------------------------------------------------------
extern "C" void kernel_launch(void* const* d_in, const int* in_sizes, int n_in,
                              void* d_out, int out_size, void* d_ws, size_t ws_size,
                              hipStream_t stream)
{
    (void)n_in; (void)out_size; (void)ws_size;

    const int NTOK = in_sizes[0];            // B*S = 4096
    const int E = 1024, Cd = 1024, H = 4096, Vv = 32000;
    const int EC = E + Cd;                   // 2048

    const int*   ids   = (const int*)d_in[0];
    const float* table = (const float*)d_in[1];
    const float* en_g  = (const float*)d_in[2];
    const float* en_b  = (const float*)d_in[3];
    const float* wA[2] = {(const float*)d_in[4],  (const float*)d_in[8]};
    const float* bA[2] = {(const float*)d_in[5],  (const float*)d_in[9]};
    const float* wB[2] = {(const float*)d_in[6],  (const float*)d_in[10]};
    const float* bB[2] = {(const float*)d_in[7],  (const float*)d_in[11]};
    const float* dw[2] = {(const float*)d_in[12], (const float*)d_in[20]};
    const float* db[2] = {(const float*)d_in[13], (const float*)d_in[21]};
    const float* fw[2] = {(const float*)d_in[14], (const float*)d_in[22]};
    const float* fb[2] = {(const float*)d_in[15], (const float*)d_in[23]};
    const float* iw[2] = {(const float*)d_in[16], (const float*)d_in[24]};
    const float* ib[2] = {(const float*)d_in[17], (const float*)d_in[25]};
    const float* cg[2] = {(const float*)d_in[18], (const float*)d_in[26]};
    const float* cb[2] = {(const float*)d_in[19], (const float*)d_in[27]};
    const float* ow    = (const float*)d_in[28];
    const float* ob    = (const float*)d_in[29];
    float* logits      = (float*)d_out;

    char*   base = (char*)d_ws;
    __bf16* xcat = (__bf16*)base;
    char*   r1   = base + (size_t)NTOK * EC * 2;   // after xcat
    __bf16* h1   = (__bf16*)r1;
    float*  g0   = (float*)r1;                      // aliases h1 (lifetimes disjoint)
    float*  g1   = g0 + (size_t)NTOK * Cd;
    float*  g2   = g1 + (size_t)NTOK * Cd;
    __bf16* h    = (__bf16*)(r1 + (size_t)3 * NTOK * Cd * 4);

    const dim3 blk(256);
    // M on x (fast dim): concurrent blocks share the same weight panel in L2.
    const dim3 gridH(NTOK / 128, H / 128);
    const dim3 gridC(NTOK / 128, Cd / 128);
    const dim3 gridV(NTOK / 128, Vv / 128);

    embed_ln_kernel<<<NTOK, blk, 0, stream>>>(ids, table, en_g, en_b, xcat);

    for (int b = 0; b < 2; ++b) {
        gemm_wmma_kernel<ACT_RELU, __bf16><<<gridH, blk, 0, stream>>>(
            xcat, EC, wA[b], H, bA[b], h1, H, EC);
        gemm_wmma_kernel<ACT_RELU, __bf16><<<gridH, blk, 0, stream>>>(
            h1, H, wB[b], H, bB[b], h, H, H);
        gemm_wmma_kernel<ACT_NONE, float><<<gridC, blk, 0, stream>>>(
            h, H, dw[b], Cd, db[b], g0, Cd, H);
        gemm_wmma_kernel<ACT_NONE, float><<<gridC, blk, 0, stream>>>(
            h, H, fw[b], Cd, fb[b], g1, Cd, H);
        gemm_wmma_kernel<ACT_NONE, float><<<gridC, blk, 0, stream>>>(
            h, H, iw[b], Cd, ib[b], g2, Cd, H);
        ctx_ln_kernel<<<NTOK, blk, 0, stream>>>(g0, g1, g2, cg[b], cb[b], xcat);
    }

    gemm_wmma_kernel<ACT_NONE, float><<<gridV, blk, 0, stream>>>(
        h, H, ow, Vv, ob, logits, Vv, H);
}
